// Decoder_19713899889333
// MI455X (gfx1250) — compile-verified, matrix operand order checked by reading, not run on hardware
//
#include <hip/hip_runtime.h>
#include <hip/hip_bf16.h>
#include <cstddef>

// ---------------------------------------------------------------------------
// MI455X (gfx1250) implementation.
//  - STFT as GEMM on v_wmma_f32_16x16x32_f16 (frames x DFT basis, K=1024)
//  - 3x3 convs as 9 shifted 1x1 GEMMs (K=32) on v_wmma_f32_16x16x32_f16:
//      channel-last activations, LDS halo tile, pre-packed f16 weight frags
//      -> every A/B fragment is one contiguous 32B load.
//  - tiny attention head on VALU (f32), ~1% of FLOPs
// Workspace use: ~56.2 MB.
// ---------------------------------------------------------------------------

typedef __attribute__((ext_vector_type(16))) _Float16 v16h;
typedef __attribute__((ext_vector_type(8)))  _Float16 v8h;
typedef __attribute__((ext_vector_type(8)))  float    v8f;

#define NF 501     // rfft bins
#define NT 401     // stft frames
#define LQ 160000  // samples per row
#define HID 250

// ------------------------------ utility ------------------------------------

__global__ void zero_out_kernel(float* __restrict__ o, int n) {
    int i = blockIdx.x * 256 + threadIdx.x;
    if (i < n) o[i] = 0.f;
}

__global__ __launch_bounds__(256) void absmax_kernel(const float* __restrict__ y,
                                                     float* __restrict__ amax_inv) {
    int b = blockIdx.x;
    float m = 0.f;
    for (int i = threadIdx.x; i < LQ; i += 256)
        m = fmaxf(m, fabsf(y[(size_t)b * LQ + i]));
    __shared__ float red[256];
    red[threadIdx.x] = m;
    __syncthreads();
    for (int s = 128; s > 0; s >>= 1) {
        if (threadIdx.x < s) red[threadIdx.x] = fmaxf(red[threadIdx.x], red[threadIdx.x + s]);
        __syncthreads();
    }
    if (threadIdx.x == 0) amax_inv[b] = 1.f / (red[0] + 1e-8f);
}

// DFT basis, TRANSPOSED: basisT[c][n], c = 2*f + (0:cos / 1:-sin), padded to 1024x1024.
__global__ void basis_kernel(_Float16* __restrict__ basisT) {
    int c = blockIdx.x;                       // 0..1023
    int n = blockIdx.y * 256 + threadIdx.x;   // 0..1023
    int f = c >> 1;
    float v = 0.f;
    if (n < 1000 && f <= 500) {
        int ph = (f * n) % 1000;              // exact modular reduction
        float th = (float)ph * 6.2831853071795864769e-3f;   // 2*pi/1000
        v = (c & 1) ? -__sinf(th) : __cosf(th);
    }
    basisT[(size_t)c * 1024 + n] = (_Float16)v;
}

// Windowed frames (identity signal), f16, [416][1024] (rows >=401 zero, cols >=1000 zero)
__global__ void frames_kernel(const float* __restrict__ y, _Float16* __restrict__ frames, int b) {
    int t = blockIdx.x;                       // 0..415
    int n = blockIdx.y * 256 + threadIdx.x;   // 0..1023
    _Float16 v = (_Float16)0.f;
    if (t < NT && n < 1000) {
        int i = t * 400 + n - 500;            // reflect pad 500
        if (i < 0) i = -i;
        if (i >= LQ) i = 2 * (LQ - 1) - i;
        float win = 0.5f * (1.0f - __cosf(6.28318530718f * (float)n * 1e-3f));
        v = (_Float16)(y[(size_t)b * LQ + i] * win);
    }
    frames[(size_t)t * 1024 + n] = v;
}

// Kmat/Vmat: Er[64][500] @ Wk^T/Wv^T + b  -> [64][250] f32
__global__ void kv_kernel(const float* __restrict__ E,
                          const float* __restrict__ Wk, const float* __restrict__ bk,
                          const float* __restrict__ Wv, const float* __restrict__ bv,
                          float* __restrict__ Kmat, float* __restrict__ Vmat) {
    int e = blockIdx.x;        // 0..63
    int q = threadIdx.x;       // 0..255
    if (q >= HID) return;
    const float* er = E + (size_t)e * 500;   // E[128][250] reshaped -> contiguous row
    float sk = bk[q], sv = bv[q];
    for (int j = 0; j < 500; ++j) {
        float x = er[j];
        sk += x * Wk[(size_t)q * 500 + j];
        sv += x * Wv[(size_t)q * 500 + j];
    }
    Kmat[e * HID + q] = sk;
    Vmat[e * HID + q] = sv;
}

// Pre-pack conv weights into WMMA A-fragments:
// wpack[mtile(2)][rs(9)][lane(32)][j(16)] f16; lane<16: rows M=lane, K 0..15;
// lane>=16: rows M=lane-16, K 16..31.  Zero-pad oc>=Cout / cin>=Cin.
__global__ void pack_w_kernel(const float* __restrict__ W, _Float16* __restrict__ wpack,
                              int Cin, int Cout) {
    int idx = blockIdx.x * 256 + threadIdx.x;
    if (idx >= 9216) return;
    int j    = idx & 15;
    int lane = (idx >> 4) & 31;
    int rs   = (idx >> 9) % 9;
    int mt   = idx / 4608;
    int kbase = (lane < 16) ? 0 : 16;
    int m   = lane & 15;
    int oc  = mt * 16 + m;
    int cin = kbase + j;
    int r = rs / 3, s = rs - r * 3;
    float v = 0.f;
    if (oc < Cout && cin < Cin)
        v = W[(((size_t)oc * Cin + cin) * 3 + r) * 3 + s];
    wpack[idx] = (_Float16)v;
}

// ------------------------------ DFT GEMM -----------------------------------
// C[t][c] = sum_n frames[t][n] * basisT[c][n]; writes both conv inputs
// channel-last: x0[sig][pix = f*401+t][ch], sig 0 = dd (scaled), sig 1 = id.
__global__ __launch_bounds__(128) void dft_kernel(const _Float16* __restrict__ frames,
                                                  const _Float16* __restrict__ basisT,
                                                  const float* __restrict__ amax_inv,
                                                  _Float16* __restrict__ x0, int b) {
    int lane = threadIdx.x & 31;
    int wave = threadIdx.x >> 5;            // wave32
    int t0 = blockIdx.x * 16;               // 26 tiles -> t < 416
    int c0 = blockIdx.y * 64 + wave * 16;   // 16 groups x 4 waves -> c < 1024
    int mrow = lane & 15;
    int kbase = (lane < 16) ? 0 : 16;

    const _Float16* arow = frames + (size_t)(t0 + mrow) * 1024 + kbase;
    const _Float16* brow = basisT + (size_t)(c0 + mrow) * 1024 + kbase;

    v8f acc = {};
    for (int ks = 0; ks < 32; ++ks) {
        v16h a  = *(const v16h*)(arow + ks * 32);   // contiguous K, 32B aligned
        v16h bb = *(const v16h*)(brow + ks * 32);
        acc = __builtin_amdgcn_wmma_f32_16x16x32_f16(false, a, false, bb,
                                                     (short)0, acc, false, false);
    }

    float inv = amax_inv[b];
    const size_t sig_stride = (size_t)NF * NT * 2;   // elems per signal
#pragma unroll
    for (int j = 0; j < 8; ++j) {
        int m = j + ((lane < 16) ? 0 : 8);
        int t = t0 + m;
        int c = c0 + (lane & 15);
        if (t < NT && c < 2 * NF) {
            int f = c >> 1, ch = c & 1;
            float v = acc[j];
            size_t idx = ((size_t)f * NT + t) * 2 + ch;
            x0[idx] = (_Float16)(v * inv);          // sig 0: dd
            x0[sig_stride + idx] = (_Float16)v;     // sig 1: identity
        }
    }
}

// --------------------------- conv as 9x 1x1 GEMM ---------------------------
// in   : f16 channel-last [pix][Cin] (Cin = 2 or 32), pix = f*401 + t
// wpack: pre-packed A fragments [2][9][32][16] f16
// out16: f16 [pix][32] (lrelu applied)  OR  out32: f32 [500][401] (oc 0, crop)
__global__ __launch_bounds__(256) void conv_kernel(const _Float16* __restrict__ in,
                                                   const _Float16* __restrict__ wpack,
                                                   const float* __restrict__ bias,
                                                   _Float16* __restrict__ out16,
                                                   float* __restrict__ out32,
                                                   int Cin, int Cout, int lrelu) {
    __shared__ alignas(64) _Float16 patch[18 * 18 * 32];   // halo tile, channel-last, 20736 B
    int t0 = blockIdx.x * 16;
    int f0 = blockIdx.y * 16;
    int tid = threadIdx.x;

    // stage halo tile: 324 pixels x 32 channels (zero-pad borders and ch >= Cin)
    for (int pix = tid; pix < 324; pix += 256) {
        int ff = pix / 18, tt = pix - ff * 18;
        int fin = f0 - 1 + ff, tin = t0 - 1 + tt;
        bool inr = (fin >= 0 && fin < NF && tin >= 0 && tin < NT);
        _Float16* dst = patch + pix * 32;
        if (Cin == 32) {
            if (inr) {
                const _Float16* src = in + ((size_t)fin * NT + tin) * 32;
                *(v16h*)(dst)      = *(const v16h*)(src);
                *(v16h*)(dst + 16) = *(const v16h*)(src + 16);
            } else {
                v16h z = {};
                *(v16h*)(dst) = z;
                *(v16h*)(dst + 16) = z;
            }
        } else {    // Cin == 2
            v16h z = {};
            if (inr) {
                const _Float16* src = in + ((size_t)fin * NT + tin) * 2;
                z[0] = src[0];
                z[1] = src[1];
            }
            v16h zz = {};
            *(v16h*)(dst) = z;
            *(v16h*)(dst + 16) = zz;
        }
    }
    __syncthreads();

    int wave = tid >> 5;
    int lane = tid & 31;
    int mtile = wave & 1;               // 16 output channels each
    int fg = wave >> 1;                 // 4 f-rows per wave
    if (mtile * 16 >= Cout) return;     // wave-uniform (convo: half the waves idle)

    int nloc = lane & 15;
    int kbase = (lane < 16) ? 0 : 16;
    int t = t0 + nloc;
    const _Float16* wbase = wpack + ((size_t)mtile * 9 * 32 + lane) * 16;

    v8f acc0 = {}, acc1 = {}, acc2 = {}, acc3 = {};
#pragma unroll
    for (int rs = 0; rs < 9; ++rs) {
        int r = rs / 3, s = rs - r * 3;
        v16h a = *(const v16h*)(wbase + rs * 512);   // one 32B load, L0-resident
        const _Float16* p = patch + ((fg * 4 + r) * 18 + (nloc + s)) * 32 + kbase;
        v16h b0 = *(const v16h*)(p);                 // fi = 0..3 rows, 32B LDS loads
        v16h b1 = *(const v16h*)(p + 1 * 18 * 32);
        v16h b2 = *(const v16h*)(p + 2 * 18 * 32);
        v16h b3 = *(const v16h*)(p + 3 * 18 * 32);
        acc0 = __builtin_amdgcn_wmma_f32_16x16x32_f16(false, a, false, b0, (short)0, acc0, false, false);
        acc1 = __builtin_amdgcn_wmma_f32_16x16x32_f16(false, a, false, b1, (short)0, acc1, false, false);
        acc2 = __builtin_amdgcn_wmma_f32_16x16x32_f16(false, a, false, b2, (short)0, acc2, false, false);
        acc3 = __builtin_amdgcn_wmma_f32_16x16x32_f16(false, a, false, b3, (short)0, acc3, false, false);
    }

    // epilogue: bias + leaky-relu, packed 16B f16 store (or cropped f32 final layer)
#pragma unroll
    for (int fi = 0; fi < 4; ++fi) {
        v8f acc = (fi == 0) ? acc0 : (fi == 1) ? acc1 : (fi == 2) ? acc2 : acc3;
        int f = f0 + fg * 4 + fi;
        if (f >= NF) continue;
        if (out16) {
            if (t < NT) {
                v8h o;
#pragma unroll
                for (int j = 0; j < 8; ++j) {
                    int oc = mtile * 16 + j + ((lane < 16) ? 0 : 8);
                    float v = acc[j] + bias[oc];
                    if (lrelu) v = (v > 0.f) ? v : 0.01f * v;
                    o[j] = (_Float16)v;
                }
                *(v8h*)(out16 + ((size_t)f * NT + t) * 32 + mtile * 16 + ((lane < 16) ? 0 : 8)) = o;
            }
        } else {
            if (t < NT && lane < 16 && f < 2 * HID)      // crop 501 -> 500, oc 0 only
                out32[(size_t)f * NT + t] = acc[0] + bias[0];
        }
    }
}

// ------------------------------- attention ---------------------------------
// One block per (chunk c, group g, batch b). ~4 GFLOP total -> VALU f32.
__global__ __launch_bounds__(256) void attend_kernel(const float* __restrict__ ex,   // [16][500][401]
                                                     const float* __restrict__ Kmat,
                                                     const float* __restrict__ Vmat,
                                                     const float* __restrict__ Wq,
                                                     const float* __restrict__ bq,
                                                     const float* __restrict__ Wdc,
                                                     const float* __restrict__ bdc,
                                                     const float* __restrict__ Wm,
                                                     const float* __restrict__ bm,
                                                     float* __restrict__ outp) {
    int c = blockIdx.x;   // 0..18
    int g = blockIdx.y;   // 0..1
    int b = blockIdx.z;   // 0..15
    __shared__ float lh[HID * 21];
    __shared__ float hbuf[16 * HID];   // h, later reused for ctx
    __shared__ float qbuf[16 * HID];
    __shared__ float lg[16 * 64];
    int tid = threadIdx.x;
    const float* exb = ex + (size_t)b * 500 * NT;

    for (int idx = tid; idx < HID * 21; idx += 256) {
        int hh = idx / 21, fu = idx % 21;
        lh[idx] = exb[(size_t)(g * HID + hh) * NT + (c * 21 + fu)];
    }
    __syncthreads();

    for (int idx = tid; idx < 16 * HID; idx += 256) {   // h = lh @ Wdc^T + bdc
        int n = idx / HID, hh = idx % HID;
        float s = bdc[n];
        for (int fu = 0; fu < 21; ++fu) s += lh[hh * 21 + fu] * Wdc[n * 21 + fu];
        hbuf[n * HID + hh] = s;
    }
    __syncthreads();

    for (int idx = tid; idx < 16 * HID; idx += 256) {   // Q = h @ Wq^T + bq
        int n = idx / HID, q = idx % HID;
        float s = bq[q];
        const float* hr = hbuf + n * HID;
        const float* wr = Wq + (size_t)q * HID;
        for (int hh = 0; hh < HID; ++hh) s += hr[hh] * wr[hh];
        qbuf[n * HID + q] = s;
    }
    __syncthreads();

    const float scale = 0.06324555320336759f;   // 1/sqrt(250)
    for (int idx = tid; idx < 16 * 64; idx += 256) {    // logits = Q @ K^T / sqrt(h)
        int n = idx / 64, k = idx % 64;
        float s = 0.f;
        const float* qr = qbuf + n * HID;
        const float* kr = Kmat + k * HID;
        for (int hh = 0; hh < HID; ++hh) s += qr[hh] * kr[hh];
        lg[idx] = s * scale;
    }
    __syncthreads();

    if (tid < 16) {                                      // softmax over k
        float mx = -1e30f;
        for (int k = 0; k < 64; ++k) mx = fmaxf(mx, lg[tid * 64 + k]);
        float sum = 0.f;
        for (int k = 0; k < 64; ++k) {
            float e = __expf(lg[tid * 64 + k] - mx);
            lg[tid * 64 + k] = e;
            sum += e;
        }
        float inv = 1.f / sum;
        for (int k = 0; k < 64; ++k) lg[tid * 64 + k] *= inv;
    }
    __syncthreads();

    for (int idx = tid; idx < 16 * HID; idx += 256) {    // ctx = elu(w @ V)
        int n = idx / HID, hh = idx % HID;
        float s = 0.f;
        for (int k = 0; k < 64; ++k) s += lg[n * 64 + k] * Vmat[k * HID + hh];
        s = (s > 0.f) ? s : (__expf(s) - 1.f);
        hbuf[n * HID + hh] = s;
    }
    __syncthreads();

    if (tid < 16) {                                      // out = ctx @ Wm + bm; mean /38
        float s = bm[0];
        for (int hh = 0; hh < HID; ++hh) s += hbuf[tid * HID + hh] * Wm[hh];
        atomicAdd(&outp[b * 16 + tid], s * (1.f / 38.f));
    }
}

// ------------------------------- launcher ----------------------------------

extern "C" void kernel_launch(void* const* d_in, const int* in_sizes, int n_in,
                              void* d_out, int out_size, void* d_ws, size_t ws_size,
                              hipStream_t stream) {
    (void)in_sizes; (void)n_in; (void)out_size; (void)ws_size;
    const float* y   = (const float*)d_in[0];
    const float* E   = (const float*)d_in[1];
    const float* wc1 = (const float*)d_in[2];
    const float* bc1 = (const float*)d_in[3];
    const float* wc2 = (const float*)d_in[4];
    const float* bc2 = (const float*)d_in[5];
    const float* wc3 = (const float*)d_in[6];
    const float* bc3 = (const float*)d_in[7];
    const float* wco = (const float*)d_in[8];
    const float* bco = (const float*)d_in[9];
    const float* Wq  = (const float*)d_in[10];
    const float* bq  = (const float*)d_in[11];
    const float* Wk  = (const float*)d_in[12];
    const float* bk  = (const float*)d_in[13];
    const float* Wv  = (const float*)d_in[14];
    const float* bv  = (const float*)d_in[15];
    const float* Wdc = (const float*)d_in[16];
    const float* bdc = (const float*)d_in[17];
    const float* Wm  = (const float*)d_in[18];
    const float* bm  = (const float*)d_in[19];

    char* ws = (char*)d_ws;
    float*     amax_inv = (float*)(ws + 0);               //      64 B
    float*     Kmat     = (float*)(ws + 4096);            //   64000 B
    float*     Vmat     = (float*)(ws + 69632);           //   64000 B
    _Float16*  wpack    = (_Float16*)(ws + 135168);       // 4 x 18432 B
    _Float16*  basisT   = (_Float16*)(ws + 212992);       // 1024*1024*2    = 2097152 B
    _Float16*  frames   = (_Float16*)(ws + 2312192);      // 416*1024*2     =  851968 B
    _Float16*  x0       = (_Float16*)(ws + 3166208);      // 2*200901*2*2   = 1607208 B
    _Float16*  a1       = (_Float16*)(ws + 4775936);      // 200901*32*2    = 12857664 B
    _Float16*  a2       = (_Float16*)(ws + 17637376);     // 200901*32*2    = 12857664 B
    float*     ex       = (float*)(ws + 30498816);        // 2*16*500*401*4 = 25664000 B
    // total ~56.2 MB

    _Float16* wp1 = wpack;
    _Float16* wp2 = wpack + 9216;
    _Float16* wp3 = wpack + 2 * 9216;
    _Float16* wpo = wpack + 3 * 9216;

    zero_out_kernel<<<2, 256, 0, stream>>>((float*)d_out, 512);
    absmax_kernel<<<16, 256, 0, stream>>>(y, amax_inv);
    basis_kernel<<<dim3(1024, 4), 256, 0, stream>>>(basisT);
    kv_kernel<<<64, 256, 0, stream>>>(E, Wk, bk, Wv, bv, Kmat, Vmat);
    pack_w_kernel<<<36, 256, 0, stream>>>(wc1, wp1, 2, 32);
    pack_w_kernel<<<36, 256, 0, stream>>>(wc2, wp2, 32, 32);
    pack_w_kernel<<<36, 256, 0, stream>>>(wc3, wp3, 32, 32);
    pack_w_kernel<<<36, 256, 0, stream>>>(wco, wpo, 32, 1);

    const size_t sigstride = (size_t)NF * NT * 2;   // x0 per-signal stride (elems)
    for (int b = 0; b < 16; ++b) {
        frames_kernel<<<dim3(416, 4), 256, 0, stream>>>(y, frames, b);
        dft_kernel<<<dim3(26, 16), 128, 0, stream>>>(frames, basisT, amax_inv, x0, b);
        for (int sig = 0; sig < 2; ++sig) {
            const _Float16* xin = x0 + (size_t)sig * sigstride;
            conv_kernel<<<dim3(26, 32), 256, 0, stream>>>(xin, wp1, bc1, a1, nullptr, 2, 32, 1);
            conv_kernel<<<dim3(26, 32), 256, 0, stream>>>(a1, wp2, bc2, a2, nullptr, 32, 32, 1);
            conv_kernel<<<dim3(26, 32), 256, 0, stream>>>(a2, wp3, bc3, a1, nullptr, 32, 32, 1);
            float* exslab = ex + ((size_t)sig * 16 + b) * 500 * NT;
            conv_kernel<<<dim3(26, 32), 256, 0, stream>>>(a1, wpo, bco, nullptr, exslab, 32, 1, 0);
        }
    }

    attend_kernel<<<dim3(19, 2, 16), 256, 0, stream>>>(
        ex, Kmat, Vmat, Wq, bq, Wdc, bdc, Wm, bm, (float*)d_out);
    attend_kernel<<<dim3(19, 2, 16), 256, 0, stream>>>(
        ex + (size_t)16 * 500 * NT, Kmat, Vmat, Wq, bq, Wdc, bdc, Wm, bm, (float*)d_out + 256);
}
